// BiLSTM_CRF_14285061226886
// MI455X (gfx1250) — matrix operand order repaired and checked by reading, not compiled, not run on hardware
//
#include <hip/hip_runtime.h>
#include <hip/hip_bf16.h>

typedef __attribute__((ext_vector_type(16))) _Float16 v16h;
typedef __attribute__((ext_vector_type(8)))  _Float16 v8h;
typedef __attribute__((ext_vector_type(8)))  float    v8f;

namespace {
constexpr int Bn = 128, Sn = 512;
constexpr int EMBn = 300, EMBp = 320;           // K padded to multiple of 32
constexpr int H2n = 256, G4n = 1024, HIDn = 512;
constexpr int TAGSn = 24, TAGSp = 32;           // N padded to 2 WMMA tiles
constexpr int TOK = Bn * Sn;                    // 65536 tokens
}

__device__ __forceinline__ float sigmoidf_(float x) { return 1.0f / (1.0f + __expf(-x)); }

// ---------------------------------------------------------------------------
// f32 -> f16 convert with zero padding (dst is R x C, valid region r0 x c0)
// ---------------------------------------------------------------------------
__global__ void pad_convert_f16(_Float16* __restrict__ dst, const float* __restrict__ src,
                                int R, int C, int r0, int c0) {
    int idx = blockIdx.x * blockDim.x + threadIdx.x;
    if (idx >= R * C) return;
    int r = idx / C, c = idx % C;
    float v = (r < r0 && c < c0) ? src[(size_t)r * c0 + c] : 0.0f;
    dst[idx] = (_Float16)v;
}

// ---------------------------------------------------------------------------
// Embedding gather -> f16, K padded 300 -> 320
// ---------------------------------------------------------------------------
__global__ void embed_gather_f16(_Float16* __restrict__ x16, const int* __restrict__ chars,
                                 const float* __restrict__ emb) {
    int tok = blockIdx.x;
    int c = threadIdx.x;                // 0..319
    int ch = chars[tok];
    float v = (c < EMBn) ? emb[(size_t)ch * EMBn + c] : 0.0f;
    x16[(size_t)tok * EMBp + c] = (_Float16)v;
}

// ---------------------------------------------------------------------------
// gx = x @ W_ih^T + b  for both directions.  f16 WMMA, f32 accumulate.
// One wave computes a 16x64 output strip.  131072 wave-units -> 16384 blocks.
// ---------------------------------------------------------------------------
__global__ void __launch_bounds__(256) gx_gemm_f16(
        _Float16* __restrict__ gx16, const _Float16* __restrict__ x16,
        const _Float16* __restrict__ wih16,
        const float* __restrict__ bias_f, const float* __restrict__ bias_b) {
    const int lane = threadIdx.x & 31;
    const int wave = threadIdx.x >> 5;
    const int unit = blockIdx.x * 8 + wave;     // 0 .. 131071
    const int dir    = unit >> 16;              // 0 fwd / 1 bwd
    const int rem    = unit & 0xFFFF;
    const int mtile  = rem >> 4;                // 0..4095 (16-token tiles)
    const int nstrip = rem & 15;                // 0..15 (64-col strips)
    const float* bias = dir ? bias_b : bias_f;
    const _Float16* wih = wih16 + (size_t)dir * G4n * EMBp;

    const _Float16* arow = x16 + (size_t)(mtile * 16 + (lane & 15)) * EMBp;
    const int akb = (lane >> 4) << 4;   // A lane K-split: {0..15} | {16..31}
    const int bkb = (lane >> 4) << 4;   // B lane K-split: {0..15} | {16..31}
    const int nl  = lane & 15;

    v8f acc[4] = {};
#pragma unroll
    for (int k0 = 0; k0 < EMBp; k0 += 32) {
        v8h lo = *(const v8h*)(arow + k0 + akb);
        v8h hi = *(const v8h*)(arow + k0 + akb + 8);
        v16h a;
#pragma unroll
        for (int i = 0; i < 8; ++i) { a[i] = lo[i]; a[i + 8] = hi[i]; }
#pragma unroll
        for (int j = 0; j < 4; ++j) {
            const int n = nstrip * 64 + j * 16 + nl;   // gate column (K-contiguous row of wih)
            v16h b = *(const v16h*)(wih + (size_t)n * EMBp + k0 + bkb);
            acc[j] = __builtin_amdgcn_wmma_f32_16x16x32_f16(false, a, false, b,
                                                            (short)0, acc[j], false, false);
        }
    }
    _Float16* gxd = gx16 + (size_t)dir * TOK * G4n;
    const int rbase = (lane < 16) ? 0 : 8;
#pragma unroll
    for (int j = 0; j < 4; ++j) {
        const int col = nstrip * 64 + j * 16 + nl;
#pragma unroll
        for (int r = 0; r < 8; ++r) {
            const int row = mtile * 16 + rbase + r;
            gxd[(size_t)row * G4n + col] = (_Float16)(acc[j][r] + bias[col]);
        }
    }
}

// ---------------------------------------------------------------------------
// BiLSTM scan.  16 workgroups: blockIdx = dir*8 + batch_slice.
// Each workgroup owns 16 batch rows and runs all 512 steps locally.
// Per step:
//   * gx[t] slice for this thread is pulled into registers BEFORE the GEMM
//     (8x16B loads, latency hidden behind 64 WMMAs)
//   * g = hA @ W_hh^T via WMMA (h lives in LDS), parked raw in gbuf
//   * gate phase: thread owns (row, 16 contiguous hidden cols); c in registers
// ---------------------------------------------------------------------------
__global__ void __launch_bounds__(256) bilstm_scan(
        const _Float16* __restrict__ gx16, const _Float16* __restrict__ whh16,
        _Float16* __restrict__ out16) {
    __shared__ __align__(64) _Float16 hA[16 * H2n];    //  8 KB: h tile (row-major, K contig)
    __shared__ __align__(64) float    gbuf[16 * G4n];  // 64 KB: pre-activation gates
    const int dir = blockIdx.x >> 3;
    const int bs  = blockIdx.x & 7;
    const int tid = threadIdx.x;
    const int lane = tid & 31;
    const int wave = tid >> 5;
    const _Float16* whh = whh16 + (size_t)dir * G4n * H2n;
    const _Float16* gxd = gx16 + (size_t)dir * TOK * G4n;

    // warm L2/WGP$ with the streaming recurrent weights (one-time sweep, 128B lines)
    for (int i = tid; i < 4096; i += 256)
        __builtin_prefetch((const char*)whh + (size_t)i * 128, 0, 3);

    for (int i = tid; i < 16 * H2n; i += 256) hA[i] = (_Float16)0.0f;
    float c[16];
#pragma unroll
    for (int k = 0; k < 16; ++k) c[k] = 0.0f;
    __syncthreads();

    const int arow  = lane & 15;
    const int akb   = (lane >> 4) << 4;
    const int bkb   = (lane >> 4) << 4;
    const int nl    = lane & 15;
    const int rbase = (lane < 16) ? 0 : 8;
    // gate-phase ownership: one batch row, 16 contiguous hidden columns
    const int grow = tid >> 4;              // 0..15
    const int hcb  = (tid & 15) * 16;       // 0,16,...,240

    for (int t = 0; t < Sn; ++t) {
        const int tt = dir ? (Sn - 1 - t) : t;
        const int btokg = (bs * 16 + grow) * Sn + tt;

        // ---- issue gx[t] register prefetch (independent of GEMM below) ----
        const _Float16* gp = gxd + (size_t)btokg * G4n;
        v8h gxr[8];
#pragma unroll
        for (int g = 0; g < 4; ++g) {
            gxr[2 * g]     = *(const v8h*)(gp + g * H2n + hcb);
            gxr[2 * g + 1] = *(const v8h*)(gp + g * H2n + hcb + 8);
        }

        // ---- recurrent GEMM: 16 x 1024 = 64 WMMA tiles, 8 per wave, K=256 ----
        v8f acc[8] = {};
#pragma unroll
        for (int k0 = 0; k0 < H2n; k0 += 32) {
            const _Float16* ap = hA + arow * H2n + k0;
            v8h lo = *(const v8h*)(ap + akb);
            v8h hi = *(const v8h*)(ap + akb + 8);
            v16h a;
#pragma unroll
            for (int i = 0; i < 8; ++i) { a[i] = lo[i]; a[i + 8] = hi[i]; }
#pragma unroll
            for (int j = 0; j < 8; ++j) {
                const int n = wave * 128 + j * 16 + nl;
                v16h b = *(const v16h*)(whh + (size_t)n * H2n + k0 + bkb);
                acc[j] = __builtin_amdgcn_wmma_f32_16x16x32_f16(false, a, false, b,
                                                                (short)0, acc[j], false, false);
            }
        }
        // ---- park raw accumulators in LDS ----
#pragma unroll
        for (int j = 0; j < 8; ++j) {
            const int col = wave * 128 + j * 16 + nl;
#pragma unroll
            for (int r = 0; r < 8; ++r) {
                gbuf[(rbase + r) * G4n + col] = acc[j][r];
            }
        }
        __syncthreads();

        // ---- gate math: add gx from registers, update c/h ----
        float hreg[16];
        const float* gb = gbuf + grow * G4n;
#pragma unroll
        for (int k = 0; k < 16; ++k) {
            const int col = hcb + k;
            const int vi = k >> 3, ei = k & 7;
            const float gi = gb[col]            + (float)gxr[0 + vi][ei];
            const float gf = gb[H2n + col]      + (float)gxr[2 + vi][ei];
            const float gg = gb[2 * H2n + col]  + (float)gxr[4 + vi][ei];
            const float go = gb[3 * H2n + col]  + (float)gxr[6 + vi][ei];
            const float cc = sigmoidf_(gf) * c[k] + sigmoidf_(gi) * tanhf(gg);
            c[k] = cc;
            hreg[k] = sigmoidf_(go) * tanhf(cc);
        }
        // vectorized h writeback: LDS (next-step A operand) + global (f16)
        v8h h0, h1;
#pragma unroll
        for (int k = 0; k < 8; ++k) { h0[k] = (_Float16)hreg[k]; h1[k] = (_Float16)hreg[k + 8]; }
        *(v8h*)(hA + grow * H2n + hcb)     = h0;
        *(v8h*)(hA + grow * H2n + hcb + 8) = h1;
        _Float16* op = out16 + (size_t)btokg * HIDn + dir * H2n + hcb;
        *(v8h*)op       = h0;
        *(v8h*)(op + 8) = h1;
        __syncthreads();
    }
}

// ---------------------------------------------------------------------------
// emissions = lstm_out @ W_out^T + b_out.  N padded 24 -> 32 (2 tiles), K=512.
// One wave per 16-token tile; 4096 waves -> 512 blocks.
// ---------------------------------------------------------------------------
__global__ void __launch_bounds__(256) emissions_gemm(
        float* __restrict__ emis, const _Float16* __restrict__ lstm16,
        const _Float16* __restrict__ wout16, const float* __restrict__ b_out) {
    const int lane = threadIdx.x & 31;
    const int wave = threadIdx.x >> 5;
    const int mtile = blockIdx.x * 8 + wave;    // 0..4095
    const _Float16* arow = lstm16 + (size_t)(mtile * 16 + (lane & 15)) * HIDn;
    const int akb = (lane >> 4) << 4;
    const int bkb = (lane >> 4) << 4;
    const int nl  = lane & 15;
    v8f acc[2] = {};
#pragma unroll
    for (int k0 = 0; k0 < HIDn; k0 += 32) {
        v8h lo = *(const v8h*)(arow + k0 + akb);
        v8h hi = *(const v8h*)(arow + k0 + akb + 8);
        v16h a;
#pragma unroll
        for (int i = 0; i < 8; ++i) { a[i] = lo[i]; a[i + 8] = hi[i]; }
#pragma unroll
        for (int j = 0; j < 2; ++j) {
            const int n = j * 16 + nl;
            v16h b = *(const v16h*)(wout16 + (size_t)n * HIDn + k0 + bkb);
            acc[j] = __builtin_amdgcn_wmma_f32_16x16x32_f16(false, a, false, b,
                                                            (short)0, acc[j], false, false);
        }
    }
    const int rbase = (lane < 16) ? 0 : 8;
#pragma unroll
    for (int j = 0; j < 2; ++j) {
        const int col = j * 16 + nl;
        if (col < TAGSn) {
#pragma unroll
            for (int r = 0; r < 8; ++r) {
                const int row = mtile * 16 + rbase + r;
                emis[(size_t)row * TAGSn + col] = acc[j][r] + b_out[col];
            }
        }
    }
}

// ---------------------------------------------------------------------------
// CRF negative log-likelihood.  One wave per batch element.
// ---------------------------------------------------------------------------
__global__ void zero_out_k(float* out) { if (threadIdx.x == 0 && blockIdx.x == 0) out[0] = 0.0f; }

__global__ void crf_nll(const float* __restrict__ emis, const int* __restrict__ tags,
                        const unsigned char* __restrict__ mask,
                        const float* __restrict__ trans, const float* __restrict__ start_trans,
                        const float* __restrict__ end_trans, float* __restrict__ out) {
    const int b = blockIdx.x;
    const int lane = threadIdx.x;
    __shared__ float alpha[TAGSn];
    __shared__ float tr[TAGSn * TAGSn];
    for (int i = lane; i < TAGSn * TAGSn; i += 32) tr[i] = trans[i];
    const float* em = emis + (size_t)b * Sn * TAGSn;
    const int* tg = tags + b * Sn;
    const unsigned char* mk = mask + b * Sn;
    if (lane < TAGSn) alpha[lane] = start_trans[lane] + em[lane];
    __syncthreads();
    for (int t = 1; t < Sn; ++t) {
        float nxt = 0.0f;
        if (lane < TAGSn) {
            float m = -3.4e38f;
#pragma unroll 4
            for (int i = 0; i < TAGSn; ++i) m = fmaxf(m, alpha[i] + tr[i * TAGSn + lane]);
            float s = 0.0f;
#pragma unroll 4
            for (int i = 0; i < TAGSn; ++i) s += __expf(alpha[i] + tr[i * TAGSn + lane] - m);
            nxt = m + __logf(s) + em[t * TAGSn + lane];
        }
        __syncthreads();
        if (lane < TAGSn && mk[t]) alpha[lane] = nxt;
        __syncthreads();
    }
    if (lane == 0) {
        float m = -3.4e38f;
        for (int i = 0; i < TAGSn; ++i) m = fmaxf(m, alpha[i] + end_trans[i]);
        float s = 0.0f;
        for (int i = 0; i < TAGSn; ++i) s += __expf(alpha[i] + end_trans[i] - m);
        const float logz = m + __logf(s);
        float score = start_trans[tg[0]] + em[tg[0]];
        int cnt = mk[0] ? 1 : 0;
        for (int t = 1; t < Sn; ++t) {
            if (mk[t]) {
                score += em[t * TAGSn + tg[t]] + tr[tg[t - 1] * TAGSn + tg[t]];
                ++cnt;
            }
        }
        score += end_trans[tg[cnt > 0 ? cnt - 1 : 0]];
        atomicAdd(out, -(score - logz) * (1.0f / (float)Bn));
    }
}

// ---------------------------------------------------------------------------
extern "C" void kernel_launch(void* const* d_in, const int* in_sizes, int n_in,
                              void* d_out, int out_size, void* d_ws, size_t ws_size,
                              hipStream_t stream) {
    (void)in_sizes; (void)n_in; (void)out_size; (void)ws_size;
    const int*   chars       = (const int*)d_in[0];
    const int*   tags        = (const int*)d_in[1];
    const unsigned char* mask = (const unsigned char*)d_in[2];
    const float* emb         = (const float*)d_in[3];
    const float* w_ih_f      = (const float*)d_in[4];
    const float* w_hh_f      = (const float*)d_in[5];
    const float* b_f         = (const float*)d_in[6];
    const float* w_ih_b      = (const float*)d_in[7];
    const float* w_hh_b      = (const float*)d_in[8];
    const float* b_b         = (const float*)d_in[9];
    const float* w_out       = (const float*)d_in[10];
    const float* b_out       = (const float*)d_in[11];
    const float* trans       = (const float*)d_in[12];
    const float* start_trans = (const float*)d_in[13];
    const float* end_trans   = (const float*)d_in[14];
    float* out = (float*)d_out;

    char* ws = (char*)d_ws;
    size_t off = 0;
    auto take = [&](size_t bytes) {
        char* p = ws + off;
        off = (off + bytes + 255) & ~(size_t)255;
        return p;
    };
    _Float16* x16    = (_Float16*)take((size_t)TOK * EMBp * 2);       //  40 MB
    _Float16* wih16  = (_Float16*)take((size_t)2 * G4n * EMBp * 2);   // 1.3 MB
    _Float16* whh16  = (_Float16*)take((size_t)2 * G4n * H2n * 2);    //   1 MB
    _Float16* wout16 = (_Float16*)take((size_t)TAGSp * HIDn * 2);     //  32 KB
    _Float16* gx16   = (_Float16*)take((size_t)2 * TOK * G4n * 2);    // 256 MB
    _Float16* lstm16 = (_Float16*)take((size_t)TOK * HIDn * 2);       //  64 MB
    float*    emis   = (float*)take((size_t)TOK * TAGSn * 4);         //   6 MB

    // weight conversion / padding to f16
    {
        int n = G4n * EMBp;
        pad_convert_f16<<<(n + 255) / 256, 256, 0, stream>>>(wih16,     w_ih_f, G4n, EMBp, G4n, EMBn);
        pad_convert_f16<<<(n + 255) / 256, 256, 0, stream>>>(wih16 + n, w_ih_b, G4n, EMBp, G4n, EMBn);
        n = G4n * H2n;
        pad_convert_f16<<<(n + 255) / 256, 256, 0, stream>>>(whh16,     w_hh_f, G4n, H2n, G4n, H2n);
        pad_convert_f16<<<(n + 255) / 256, 256, 0, stream>>>(whh16 + n, w_hh_b, G4n, H2n, G4n, H2n);
        n = TAGSp * HIDn;
        pad_convert_f16<<<(n + 255) / 256, 256, 0, stream>>>(wout16, w_out, TAGSp, HIDn, TAGSn, HIDn);
    }
    embed_gather_f16<<<TOK, EMBp, 0, stream>>>(x16, chars, emb);
    gx_gemm_f16<<<16384, 256, 0, stream>>>(gx16, x16, wih16, b_f, b_b);
    bilstm_scan<<<16, 256, 0, stream>>>(gx16, whh16, lstm16);
    emissions_gemm<<<512, 256, 0, stream>>>(emis, lstm16, wout16, b_out);
    zero_out_k<<<1, 1, 0, stream>>>(out);
    crf_nll<<<Bn, 32, 0, stream>>>(emis, tags, mask, trans, start_trans, end_trans, out);
}
